// MotionPrediction_39324720562688
// MI455X (gfx1250) — matrix operand order repaired and checked by reading, not compiled, not run on hardware
//
#include <hip/hip_runtime.h>
#include <hip/hip_bf16.h>

typedef __attribute__((ext_vector_type(16))) __bf16  v16bf;
typedef __attribute__((ext_vector_type(8)))  float   v8f;
typedef __attribute__((ext_vector_type(4)))  float   f32x4;
typedef __attribute__((ext_vector_type(4)))  __bf16  bf16x4;

// Problem constants (match reference)
#define NB   65536
#define NF   256      // IN == HID == OUT == 256

// 32-byte WMMA fragment viewed as two 16B chunks for b128 loads
union Frag {
    v16bf v;
    f32x4 f4[2];
};

// ---------------------------------------------------------------------------
// Per-sample expert-blend coefficients: C[b][e] (scatter coeff[i] -> k[i])
// ---------------------------------------------------------------------------
__global__ __launch_bounds__(256) void coeff_kernel(const float* __restrict__ phi,
                                                    float* __restrict__ C) {
    int b = blockIdx.x * blockDim.x + threadIdx.x;
    if (b >= NB) return;
    float w  = phi[b] * 0.6366197723675814f;   // 2/pi
    int  wi  = ((int)w) & 3;                    // w >= 0, trunc == floor
    float w2 = w * w, w3 = w2 * w;
    float c0 = -0.5f * w + w2 - 0.5f * w3;
    float c1 = -2.5f * w2 + 1.5f * w3;
    float c2 =  0.5f * w + 2.0f * w2 - 1.5f * w3;
    float c3 = -0.5f * w2 + 0.5f * w3;
    float* cr = C + b * 4;
    cr[(wi + 3) & 3] = c0;
    cr[ wi         ] = c1;
    cr[(wi + 1) & 3] = c2;
    cr[(wi + 2) & 3] = c3;
}

// ---------------------------------------------------------------------------
// f32 -> bf16 convert (vectorized 4/thread)
// ---------------------------------------------------------------------------
__global__ __launch_bounds__(256) void cvt_bf16_kernel(const float* __restrict__ x,
                                                       __bf16* __restrict__ y, int n4) {
    int i = blockIdx.x * blockDim.x + threadIdx.x;
    if (i >= n4) return;
    f32x4 v = *reinterpret_cast<const f32x4*>(x + i * 4);
    bf16x4 o;
    o.x = (__bf16)v.x; o.y = (__bf16)v.y; o.z = (__bf16)v.z; o.w = (__bf16)v.w;
    *reinterpret_cast<bf16x4*>(y + i * 4) = o;
}

// ---------------------------------------------------------------------------
// W[e][k][n] f32  ->  Wt[e][n][k] bf16   (4 x 256 x 256)
// ---------------------------------------------------------------------------
__global__ __launch_bounds__(256) void wtrans_kernel(const float* __restrict__ W,
                                                     __bf16* __restrict__ Wt) {
    int idx = blockIdx.x * blockDim.x + threadIdx.x;   // (e*256+n)*256+k
    if (idx >= 4 * NF * NF) return;
    int k = idx & 255;
    int n = (idx >> 8) & 255;
    int e = idx >> 16;
    Wt[idx] = (__bf16)W[(e << 16) + (k << 8) + n];
}

// ---------------------------------------------------------------------------
// One MoE layer: out[b,n] = act( sum_e C[b][e] * ( Xin[b,:] @ Wt[e][n,:] + bias[e][n] ) )
// Block: 256 thr (8 waves). Block tile: M=256 (32 rows/wave) x N=32 x 4 experts, K=256.
// Weights double-buffered through LDS in K-chunks of 64; 1 barrier per chunk.
// ---------------------------------------------------------------------------
template <bool RELU, bool OUTF32>
__global__ __launch_bounds__(256) void layer_kernel(const __bf16* __restrict__ Xin,
                                                    const __bf16* __restrict__ Wt,
                                                    const float*  __restrict__ bias,
                                                    const float*  __restrict__ C,
                                                    void* __restrict__ outp) {
    constexpr int K       = NF;
    constexpr int LSTRIDE = 72;                       // 64 K-elems + 8 pad (bank-friendly)
    constexpr int LBUF    = 4 * 32 * LSTRIDE;         // one stage buffer (elems)
    __shared__ __bf16 lds[2 * LBUF];                  // double buffer (~36 KB)

    const int lane = threadIdx.x & 31;
    const int wave = threadIdx.x >> 5;
    const int m0   = blockIdx.x * 256 + wave * 32;    // 32 rows per wave
    const int n0   = blockIdx.y * 32;
    const int hi   = lane >> 4;                       // half-wave select
    const int l15  = lane & 15;

    // acc[e][t][ms] : expert e, n-tile t, m-subtile ms
    v8f acc[4][2][2];
#pragma unroll
    for (int e = 0; e < 4; ++e)
#pragma unroll
        for (int t = 0; t < 2; ++t)
#pragma unroll
            for (int ms = 0; ms < 2; ++ms) acc[e][t][ms] = 0.0f;

    const __bf16* arow0 = Xin + (m0 + l15) * K;       // m-subtile 0
    const __bf16* arow1 = arow0 + 16 * K;             // m-subtile 1

    // per-thread weight staging addresses (16B granules)
    const __bf16* gptr[4];
    int           lofs[4];
#pragma unroll
    for (int j = 0; j < 4; ++j) {
        int cid = threadIdx.x + j * 256;              // 0..1023
        int row = cid >> 3;                           // e*32 + nl, 0..127
        int off = (cid & 7) * 8;                      // element offset
        int e   = row >> 5;
        int nl  = row & 31;
        gptr[j] = Wt + ((e * NF + n0 + nl) * K + off);
        lofs[j] = row * LSTRIDE + off;
    }

    // prologue: stage chunk 0 into buffer 0
    {
        f32x4 st[4];
#pragma unroll
        for (int j = 0; j < 4; ++j) st[j] = *reinterpret_cast<const f32x4*>(gptr[j]);
#pragma unroll
        for (int j = 0; j < 4; ++j) *reinterpret_cast<f32x4*>(&lds[lofs[j]]) = st[j];
    }

    for (int kc = 0; kc < 4; ++kc) {                  // K chunks of 64
        const int  cur     = kc & 1;
        const bool hasNext = (kc < 3);

        // issue next chunk's global loads (overlap with barrier + compute)
        f32x4 st[4];
        if (hasNext) {
            const int kn = (kc + 1) * 64;
#pragma unroll
            for (int j = 0; j < 4; ++j)
                st[j] = *reinterpret_cast<const f32x4*>(gptr[j] + kn);
            __builtin_prefetch(arow0 + kn, 0, 1);     // global_prefetch_b8
            __builtin_prefetch(arow1 + kn, 0, 1);
        }

        __syncthreads();                              // buf[cur] ready; buf[1-cur] free

        const __bf16* lbase = &lds[cur * LBUF];
#pragma unroll
        for (int ks = 0; ks < 64; ks += 32) {
            // A fragments: 16x32 bf16. lane<16: K{0..7,16..23}; lane>=16: +8
            Frag a0, a1;
            const int ak = kc * 64 + ks + hi * 8;
            a0.f4[0] = *reinterpret_cast<const f32x4*>(arow0 + ak);
            a0.f4[1] = *reinterpret_cast<const f32x4*>(arow0 + ak + 16);
            a1.f4[0] = *reinterpret_cast<const f32x4*>(arow1 + ak);
            a1.f4[1] = *reinterpret_cast<const f32x4*>(arow1 + ak + 16);

#pragma unroll
            for (int e = 0; e < 4; ++e) {
#pragma unroll
                for (int t = 0; t < 2; ++t) {
                    // B fragment: 32x16. lane<16: col n K{0..7,16..23}; lane>=16: +8
                    Frag b;
                    const __bf16* lp =
                        lbase + (e * 32 + t * 16 + l15) * LSTRIDE + ks + hi * 8;
                    b.f4[0] = *reinterpret_cast<const f32x4*>(lp);
                    b.f4[1] = *reinterpret_cast<const f32x4*>(lp + 16);
                    acc[e][t][0] = __builtin_amdgcn_wmma_f32_16x16x32_bf16(
                        false, a0.v, false, b.v, (short)0, acc[e][t][0], false, false);
                    acc[e][t][1] = __builtin_amdgcn_wmma_f32_16x16x32_bf16(
                        false, a1.v, false, b.v, (short)0, acc[e][t][1], false, false);
                }
            }
        }

        // store next chunk into the other buffer (its readers finished pre-barrier)
        if (hasNext) {
            __bf16* dbase = &lds[(1 - cur) * LBUF];
#pragma unroll
            for (int j = 0; j < 4; ++j)
                *reinterpret_cast<f32x4*>(dbase + lofs[j]) = st[j];
        }
    }

    // ---- epilogue: bias + per-row expert blend (+ReLU) ----
#pragma unroll
    for (int ms = 0; ms < 2; ++ms) {
        const int mb = m0 + ms * 16;
        f32x4 cv[8];
#pragma unroll
        for (int r = 0; r < 8; ++r) {
            int row = mb + r + hi * 8;
            cv[r] = *reinterpret_cast<const f32x4*>(C + row * 4);
        }
#pragma unroll
        for (int t = 0; t < 2; ++t) {
            int col = n0 + t * 16 + l15;
            float b0 = bias[         col];
            float b1 = bias[    NF + col];
            float b2 = bias[2 * NF + col];
            float b3 = bias[3 * NF + col];
#pragma unroll
            for (int r = 0; r < 8; ++r) {
                int row = mb + r + hi * 8;
                float v = cv[r].x * (acc[0][t][ms][r] + b0)
                        + cv[r].y * (acc[1][t][ms][r] + b1)
                        + cv[r].z * (acc[2][t][ms][r] + b2)
                        + cv[r].w * (acc[3][t][ms][r] + b3);
                if (RELU) v = fmaxf(v, 0.0f);
                if (OUTF32)
                    reinterpret_cast<float*>(outp)[row * NF + col] = v;
                else
                    reinterpret_cast<__bf16*>(outp)[row * NF + col] = (__bf16)v;
            }
        }
    }
}

// ---------------------------------------------------------------------------
extern "C" void kernel_launch(void* const* d_in, const int* in_sizes, int n_in,
                              void* d_out, int out_size, void* d_ws, size_t ws_size,
                              hipStream_t stream) {
    const float* X   = (const float*)d_in[0];
    const float* phi = (const float*)d_in[1];
    const float* W1  = (const float*)d_in[2];
    const float* b1  = (const float*)d_in[3];
    const float* W2  = (const float*)d_in[4];
    const float* b2  = (const float*)d_in[5];
    const float* W3  = (const float*)d_in[6];
    const float* b3  = (const float*)d_in[7];

    char* ws = (char*)d_ws;
    const size_t actBytes = (size_t)NB * NF * sizeof(__bf16);   // 32 MB
    __bf16* Xh  = (__bf16*)(ws);                  // X bf16; later reused as H2
    __bf16* H1  = (__bf16*)(ws + actBytes);       // H1 bf16
    __bf16* Wt1 = (__bf16*)(ws + 2 * actBytes);
    __bf16* Wt2 = Wt1 + 4 * NF * NF;
    __bf16* Wt3 = Wt2 + 4 * NF * NF;
    float*  C   = (float*)((char*)(Wt3 + 4 * NF * NF));

    // prep
    coeff_kernel<<<NB / 256, 256, 0, stream>>>(phi, C);
    {
        int n4 = NB * NF / 4;
        cvt_bf16_kernel<<<(n4 + 255) / 256, 256, 0, stream>>>(X, Xh, n4);
    }
    wtrans_kernel<<<(4 * NF * NF) / 256, 256, 0, stream>>>(W1, Wt1);
    wtrans_kernel<<<(4 * NF * NF) / 256, 256, 0, stream>>>(W2, Wt2);
    wtrans_kernel<<<(4 * NF * NF) / 256, 256, 0, stream>>>(W3, Wt3);

    // 3 fused expert-GEMM + blend layers
    dim3 grid(NB / 256, NF / 32, 1);
    layer_kernel<true,  false><<<grid, 256, 0, stream>>>(Xh, Wt1, b1, C, H1);
    layer_kernel<true,  false><<<grid, 256, 0, stream>>>(H1, Wt2, b2, C, Xh);   // H2 -> Xh slot
    layer_kernel<false, true ><<<grid, 256, 0, stream>>>(Xh, Wt3, b3, C, d_out);

    (void)in_sizes; (void)n_in; (void)out_size; (void)ws_size;
}
